// PINN_70274254897241
// MI455X (gfx1250) — matrix operand order repaired and checked by reading, not compiled, not run on hardware
//
#include <hip/hip_runtime.h>
#include <hip/hip_bf16.h>
#include <stdint.h>

// ---------------------------------------------------------------------------
// PINN Navier-Stokes forward pass for gfx1250 (MI455X).
// 13 derivative streams propagated through 3 hidden 256x256 GEMMs via
// v_wmma_f32_16x16x32_f16; tanh chain rule fused per-lane between layers.
// ---------------------------------------------------------------------------

typedef __attribute__((ext_vector_type(16))) _Float16 v16h;
typedef __attribute__((ext_vector_type(8)))  float    v8f;
typedef __attribute__((ext_vector_type(4)))  uint32_t v4u;

#define HID   256
#define PTS   16            // points per workgroup
#define NSTR  13            // derivative streams
#define MROWS (NSTR * PTS)  // 208 rows = 13 WMMA M-tiles
#define AS    264           // LDS row stride in halves (256 + 8 pad)

union AFrag { v16h h; v4u u4[2]; };

// Convert W2/W3/W4 (f32, [k][n]) -> transposed f16 WT[layer][n][k] in d_ws.
__global__ void pinn_prep_weights(const float* __restrict__ W2,
                                  const float* __restrict__ W3,
                                  const float* __restrict__ W4,
                                  _Float16* __restrict__ WT) {
  int idx = blockIdx.x * blockDim.x + threadIdx.x;   // 0 .. 3*65536-1
  int l = idx >> 16;
  int r = idx & 0xFFFF;
  int n = r >> 8;          // output unit
  int k = r & 255;         // input unit
  const float* W = (l == 0) ? W2 : ((l == 1) ? W3 : W4);
  WT[(size_t)l * HID * HID + n * HID + k] = (_Float16)W[k * HID + n];
}

__global__ __launch_bounds__(256) void pinn_forward_kernel(
    const float* __restrict__ xin, const float* __restrict__ yin,
    const float* __restrict__ tin,
    const float* __restrict__ W1, const float* __restrict__ b1,
    const _Float16* __restrict__ WT,   // 3 x [256][256] transposed f16
    const float* __restrict__ b2, const float* __restrict__ b3,
    const float* __restrict__ b4,
    const float* __restrict__ W5, const float* __restrict__ b5,
    const float* __restrict__ lam1p, const float* __restrict__ lam2p,
    float* __restrict__ out, int N)
{
  __shared__ __align__(16) _Float16 Abuf[2][MROWS * AS];  // ping-pong streams
  __shared__ float res[PTS][26];                          // final dots

  const int tid  = threadIdx.x;
  const int lane = tid & 31;
  const int wid  = tid >> 5;           // 8 waves
  const int basept = blockIdx.x * PTS;

  // ---------------- layer 1: q -> h1 streams (analytic, tiny) -------------
  for (int i = tid; i < PTS * HID; i += 256) {
    int p = i >> 8;
    int j = i & 255;
    int gi = basept + p; if (gi >= N) gi = N - 1;
    float qx = xin[gi], qy = yin[gi], qt = tin[gi];
    float wx = W1[0 * HID + j], wy = W1[1 * HID + j], wt = W1[2 * HID + j];
    float z  = qx * wx + qy * wy + qt * wt + b1[j];
    float u  = tanhf(z);
    float d1 = 1.f - u * u;            // sigma'
    float d2 = -2.f * u * d1;          // sigma''
    float d3 = -2.f * (d1 * d1 + u * d2); // sigma'''
    _Float16* A = Abuf[0];
    float vals[NSTR] = {
      u,
      d1 * wx, d1 * wy, d1 * wt,
      d2 * wx * wx, d2 * wx * wy, d2 * wy * wy, d2 * wx * wt, d2 * wy * wt,
      d3 * wx * wx * wx, d3 * wx * wx * wy, d3 * wx * wy * wy, d3 * wy * wy * wy
    };
    #pragma unroll
    for (int s = 0; s < NSTR; ++s)
      A[(s * PTS + p) * AS + j] = (_Float16)vals[s];
  }

  // ---------------- layers 2..4: WMMA GEMM + fused tanh chain rule --------
  int cur = 0;
  const int hi = lane >> 4;            // half-wave select (0/1)
  for (int layer = 0; layer < 3; ++layer) {
    __syncthreads();
    const _Float16* A  = Abuf[cur];
    _Float16*       An = Abuf[cur ^ 1];
    const _Float16* W  = WT + (size_t)layer * HID * HID;
    const float* bias  = (layer == 0) ? b2 : ((layer == 1) ? b3 : b4);

    for (int nt = wid; nt < 16; nt += 8) {        // N-tiles, uniform per wave
      const int ncol = nt * 16 + (lane & 15);     // this lane's output unit
      const float bv = bias[ncol];

      v8f acc[NSTR];
      #pragma unroll
      for (int s = 0; s < NSTR; ++s)
        acc[s] = (v8f){0.f,0.f,0.f,0.f,0.f,0.f,0.f,0.f};

      #pragma unroll
      for (int kc = 0; kc < 8; ++kc) {            // K = 256 in chunks of 32
        const int kb = kc * 32;
        // B fragment (32x16 f16): lanes0-15 K=kb..kb+15, lanes16-31 next 16
        AFrag bf;
        const v4u* bp = (const v4u*)(W + (size_t)ncol * HID + kb + hi * 16);
        bf.u4[0] = bp[0];
        bf.u4[1] = bp[1];
        #pragma unroll
        for (int s = 0; s < NSTR; ++s) {
          // A fragment (16x32 f16): row = stream tile row; lanes0-15 K=kb..kb+15,
          // lanes16-31 next 16 (contiguous, matches B)
          const int row = s * PTS + (lane & 15);
          AFrag af;
          af.u4[0] = *(const v4u*)&A[row * AS + kb + hi * 16];
          af.u4[1] = *(const v4u*)&A[row * AS + kb + hi * 16 + 8];
          acc[s] = __builtin_amdgcn_wmma_f32_16x16x32_f16(
              false, af.h, false, bf.h, (short)0, acc[s], false, false);
        }
      }

      // tanh chain rule: all 13 streams for (point,unit) sit in the same
      // lane / same C-slot across the 13 accumulators.
      #pragma unroll
      for (int r = 0; r < 8; ++r) {
        const int p = r + hi * 8;                 // C layout: M = r (+8)
        float z0 = acc[0][r] + bv;
        float u  = tanhf(z0);
        float d1 = 1.f - u * u;
        float d2 = -2.f * u * d1;
        float d3 = -2.f * (d1 * d1 + u * d2);
        float zx  = acc[1][r],  zy  = acc[2][r],  zt  = acc[3][r];
        float zxx = acc[4][r],  zxy = acc[5][r],  zyy = acc[6][r];
        float zxt = acc[7][r],  zyt = acc[8][r];
        float zxxx = acc[9][r], zxxy = acc[10][r];
        float zxyy = acc[11][r], zyyy = acc[12][r];
        float vals[NSTR] = {
          u,
          d1 * zx, d1 * zy, d1 * zt,
          d2 * zx * zx + d1 * zxx,
          d2 * zx * zy + d1 * zxy,
          d2 * zy * zy + d1 * zyy,
          d2 * zx * zt + d1 * zxt,
          d2 * zy * zt + d1 * zyt,
          d3 * zx * zx * zx + 3.f * d2 * zxx * zx + d1 * zxxx,
          d3 * zx * zx * zy + d2 * (zxx * zy + 2.f * zxy * zx) + d1 * zxxy,
          d3 * zx * zy * zy + d2 * (zyy * zx + 2.f * zxy * zy) + d1 * zxyy,
          d3 * zy * zy * zy + 3.f * d2 * zyy * zy + d1 * zyyy
        };
        #pragma unroll
        for (int s = 0; s < NSTR; ++s)
          An[(s * PTS + p) * AS + ncol] = (_Float16)vals[s];
      }
    }
    cur ^= 1;
  }

  // ---------------- layer 5 (256 -> 2, linear) + NS residual assembly -----
  __syncthreads();
  const _Float16* A = Abuf[cur];
  for (int idx = tid; idx < PTS * 26; idx += 256) {
    int p  = idx / 26;
    int sc = idx % 26;
    int s = sc >> 1, c = sc & 1;
    const _Float16* row = A + (s * PTS + p) * AS;
    float sum = 0.f;
    for (int k = 0; k < HID; ++k)
      sum += (float)row[k] * W5[k * 2 + c];
    res[p][sc] = sum;
  }
  __syncthreads();
  if (tid < PTS) {
    const int p = tid;
    const float lam1 = lam1p[0], lam2 = lam2p[0];
    float p_pred = res[p][0 * 2 + 1] + b5[1];
    float sx = res[p][1 * 2 + 0], sy = res[p][2 * 2 + 0];
    float px = res[p][1 * 2 + 1], py = res[p][2 * 2 + 1];
    float sxx = res[p][4 * 2 + 0], sxy = res[p][5 * 2 + 0];
    float syy = res[p][6 * 2 + 0];
    float sxt = res[p][7 * 2 + 0], syt = res[p][8 * 2 + 0];
    float sxxx = res[p][9 * 2 + 0],  sxxy = res[p][10 * 2 + 0];
    float sxyy = res[p][11 * 2 + 0], syyy = res[p][12 * 2 + 0];

    float u = sy, v = -sx;
    float u_t = syt,  v_t = -sxt;
    float u_x = sxy,  v_x = -sxx;
    float u_y = syy,  v_y = -sxy;
    float u_xx = sxxy, v_xx = -sxxx;
    float u_yy = syyy, v_yy = -sxyy;
    float f = lam1 * (u_t + u * u_x + v * u_y) + px - lam2 * (u_xx + u_yy);
    float g = lam1 * (v_t + u * v_x + v * v_y) + py - lam2 * (v_xx + v_yy);

    int gi = basept + p;
    if (gi < N) {
      out[0 * N + gi] = p_pred;
      out[1 * N + gi] = u;
      out[2 * N + gi] = v;
      out[3 * N + gi] = f;
      out[4 * N + gi] = g;
    }
  }
}

extern "C" void kernel_launch(void* const* d_in, const int* in_sizes, int n_in,
                              void* d_out, int out_size, void* d_ws, size_t ws_size,
                              hipStream_t stream) {
  // input order: x,y,t,p,u,v,W1,b1,W2,b2,W3,b3,W4,b4,W5,b5,lambda1,lambda2
  const float* x  = (const float*)d_in[0];
  const float* y  = (const float*)d_in[1];
  const float* t  = (const float*)d_in[2];
  const float* W1 = (const float*)d_in[6];
  const float* b1 = (const float*)d_in[7];
  const float* W2 = (const float*)d_in[8];
  const float* b2 = (const float*)d_in[9];
  const float* W3 = (const float*)d_in[10];
  const float* b3 = (const float*)d_in[11];
  const float* W4 = (const float*)d_in[12];
  const float* b4 = (const float*)d_in[13];
  const float* W5 = (const float*)d_in[14];
  const float* b5 = (const float*)d_in[15];
  const float* l1 = (const float*)d_in[16];
  const float* l2 = (const float*)d_in[17];
  float* out = (float*)d_out;
  int N = in_sizes[0];

  _Float16* WT = (_Float16*)d_ws;   // 3*256*256 f16 = 384 KB scratch

  pinn_prep_weights<<<(3 * HID * HID) / 256, 256, 0, stream>>>(W2, W3, W4, WT);
  pinn_forward_kernel<<<(N + PTS - 1) / PTS, 256, 0, stream>>>(
      x, y, t, W1, b1, WT, b2, b3, b4, W5, b5, l1, l2, out, N);
}
